// unit_vit_22454089024040
// MI455X (gfx1250) — matrix operand order repaired and checked by reading, not compiled, not run on hardware
//
#include <hip/hip_runtime.h>
#include <hip/hip_bf16.h>

typedef __attribute__((ext_vector_type(16))) _Float16 v16h;
typedef __attribute__((ext_vector_type(8)))  _Float16 v8h;
typedef __attribute__((ext_vector_type(8)))  float    v8f;

#define VJ   25
#define NEH  5
#define DIMC 192
#define HH   6
#define HD   32
#define NB   64
#define TT   64
#define ROWS (NB * TT * VJ)          // 102400
#define SCALE_F 0.17677669529663687f // 32^-0.5
#define COEFF_F 0.4f

// ---------------- prep kernels ----------------

__global__ void prep_hadj_kernel(const int* __restrict__ jl,
                                 const float* __restrict__ he,
                                 float* __restrict__ h_adj) {
    __shared__ int cnt[NEH];
    int tid = threadIdx.x;
    if (tid < NEH) cnt[tid] = 0;
    __syncthreads();
    if (tid < VJ) atomicAdd(&cnt[jl[tid]], 1);
    __syncthreads();
    for (int idx = tid; idx < VJ * VJ; idx += blockDim.x) {
        int v = idx / VJ, w = idx % VJ;
        int lv = jl[v], lw = jl[w];
        h_adj[idx] = (lv == lw) ? (he[lv] / (float)cnt[lv]) : 0.0f;
    }
}

__global__ void f2h_kernel(const float* __restrict__ src, _Float16* __restrict__ dst, int n) {
    int i = blockIdx.x * blockDim.x + threadIdx.x;
    if (i < n) dst[i] = (_Float16)src[i];
}

// build block-diagonal 192x192 (f16) from grouped proj_w (192x32)
__global__ void wbd_kernel(const float* __restrict__ proj_w, _Float16* __restrict__ wbd) {
    int i = blockIdx.x * blockDim.x + threadIdx.x;
    if (i >= DIMC * DIMC) return;
    int o = i / DIMC, ic = i % DIMC;
    wbd[i] = (o / HD == ic / HD) ? (_Float16)proj_w[o * HD + (ic % HD)] : (_Float16)0.0f;
}

// ---------------- stage kernels ----------------

// z[row=(n,t,w)][c] = sum_v x[n,c,t,v] * h_adj[v,w]   (f16 out)
__global__ void z_kernel(const float* __restrict__ x, const float* __restrict__ h_adj,
                         _Float16* __restrict__ zbuf) {
    __shared__ float xt[DIMC * VJ];
    __shared__ float ha[VJ * VJ];
    int nt = blockIdx.x;                 // n*TT + t
    int n = nt >> 6, t = nt & 63;
    int tid = threadIdx.x;
    for (int idx = tid; idx < DIMC * VJ; idx += blockDim.x) {
        int c = idx / VJ, v = idx % VJ;
        xt[idx] = x[(((size_t)n * DIMC + c) * TT + t) * VJ + v];
    }
    for (int idx = tid; idx < VJ * VJ; idx += blockDim.x) ha[idx] = h_adj[idx];
    __syncthreads();
    for (int idx = tid; idx < VJ * DIMC; idx += blockDim.x) {
        int w = idx / DIMC, c = idx % DIMC;
        float acc = 0.0f;
        #pragma unroll
        for (int v = 0; v < VJ; ++v) acc += xt[c * VJ + v] * ha[v * VJ + w];
        zbuf[((size_t)nt * VJ + w) * DIMC + c] = (_Float16)acc;
    }
}

// LayerNorm over channels, f16 rows out
__global__ void ln_kernel(const float* __restrict__ x, const float* __restrict__ g,
                          const float* __restrict__ b, _Float16* __restrict__ xn) {
    __shared__ float xt[DIMC * VJ];
    __shared__ float mu[VJ], rs[VJ];
    int nt = blockIdx.x;
    int n = nt >> 6, t = nt & 63;
    int tid = threadIdx.x;
    for (int idx = tid; idx < DIMC * VJ; idx += blockDim.x) {
        int c = idx / VJ, v = idx % VJ;
        xt[idx] = x[(((size_t)n * DIMC + c) * TT + t) * VJ + v];
    }
    __syncthreads();
    if (tid < VJ) {
        float s = 0.0f, s2 = 0.0f;
        for (int c = 0; c < DIMC; ++c) { float vv = xt[c * VJ + tid]; s += vv; s2 += vv * vv; }
        float m = s / (float)DIMC;
        mu[tid] = m;
        float var = s2 / (float)DIMC - m * m;
        rs[tid] = rsqrtf(var + 1e-5f);
    }
    __syncthreads();
    for (int idx = tid; idx < VJ * DIMC; idx += blockDim.x) {
        int v = idx / DIMC, c = idx % DIMC;
        float val = (xt[c * VJ + v] - mu[v]) * rs[v] * g[c] + b[c];
        xn[((size_t)nt * VJ + v) * DIMC + c] = (_Float16)val;
    }
}

// WMMA GEMM: C[M x Ncol] = A[M x K] * B^T  (B given as [Ncol x K] row-major)
// out_half selects f16 vs f32 output.  One wave per 16x16 tile.
__global__ void gemm_wmma_kernel(const _Float16* __restrict__ A,
                                 const _Float16* __restrict__ B,
                                 float* __restrict__ Cf, _Float16* __restrict__ Ch,
                                 int M, int K, int Ncol, int out_half) {
    int wid  = (int)((blockIdx.x * blockDim.x + threadIdx.x) >> 5);
    int lane = threadIdx.x & 31;
    int ntiles = Ncol >> 4;
    int mtile = wid / ntiles, ntile = wid % ntiles;
    if (mtile * 16 >= M) return;
    int g  = lane >> 4;
    int mr = lane & 15;
    const _Float16* arow = A + (size_t)(mtile * 16 + mr) * K + 16 * g;
    const _Float16* brow = B + (size_t)(ntile * 16 + mr) * K + 16 * g;
    v8f acc = {};
    for (int kk = 0; kk < K; kk += 32) {
        v16h a = *(const v16h*)(arow + kk);        // K = kk+16g .. +15 (contiguous)
        v16h bfrag = *(const v16h*)(brow + kk);    // K = kk+16g .. +15 (contiguous)
        acc = __builtin_amdgcn_wmma_f32_16x16x32_f16(false, a, false, bfrag,
                                                     (short)0, acc, false, false);
    }
    int n = ntile * 16 + mr;
    int mbase = mtile * 16 + 8 * g;
    if (out_half) {
        #pragma unroll
        for (int r = 0; r < 8; ++r) Ch[(size_t)(mbase + r) * Ncol + n] = (_Float16)acc[r];
    } else {
        #pragma unroll
        for (int r = 0; r < 8; ++r) Cf[(size_t)(mbase + r) * Ncol + n] = acc[r];
    }
}

// s[n,t] = mean over (v,c) of op rows (4800 contiguous floats per (n,t))
__global__ void s_kernel(const float* __restrict__ op, float* __restrict__ s) {
    __shared__ float red[256];
    int nt = blockIdx.x;
    const float* p = op + (size_t)nt * VJ * DIMC;
    float a = 0.0f;
    for (int i = threadIdx.x; i < VJ * DIMC; i += 256) a += p[i];
    red[threadIdx.x] = a;
    __syncthreads();
    for (int off = 128; off > 0; off >>= 1) {
        if (threadIdx.x < off) red[threadIdx.x] += red[threadIdx.x + off];
        __syncthreads();
    }
    if (threadIdx.x == 0) s[nt] = red[0] / (float)(VJ * DIMC);
}

// squeeze-excite MLP per sample n: 64 -> 12 (relu) -> 64 (sigmoid)
__global__ void g_kernel(const float* __restrict__ s,
                         const float* __restrict__ w1, const float* __restrict__ b1,
                         const float* __restrict__ w2, const float* __restrict__ b2,
                         float* __restrict__ g) {
    __shared__ float sv[TT], h1[TT / 5];
    int n = blockIdx.x, tid = threadIdx.x;
    sv[tid] = s[n * TT + tid];
    __syncthreads();
    if (tid < TT / 5) {
        float a = b1[tid];
        for (int t = 0; t < TT; ++t) a += sv[t] * w1[tid * TT + t];
        h1[tid] = fmaxf(a, 0.0f);
    }
    __syncthreads();
    float a = b2[tid];
    for (int j = 0; j < TT / 5; ++j) a += h1[j] * w2[tid * (TT / 5) + j];
    g[n * TT + tid] = 1.0f / (1.0f + expf(-a));
}

// e = op * (1 + COEFF * g[n,t])  -> f16 rows
__global__ void e_kernel(const float* __restrict__ op, const float* __restrict__ g,
                         _Float16* __restrict__ e, size_t total) {
    size_t i = (size_t)blockIdx.x * blockDim.x + threadIdx.x;
    if (i >= total) return;
    size_t row = i / DIMC;
    int nt = (int)(row / VJ);
    e[i] = (_Float16)(op[i] * (1.0f + COEFF_F * g[nt]));
}

// attention: one wave per (b,t,h); 4 waves per block
__global__ void attn_kernel(const _Float16* __restrict__ qb, const _Float16* __restrict__ kvb,
                            const _Float16* __restrict__ eb, const int* __restrict__ hops,
                            const float* __restrict__ rpe, const float* __restrict__ w1,
                            const float* __restrict__ outer, const float* __restrict__ alpha_p,
                            _Float16* __restrict__ ob, int n_hop) {
    __shared__ _Float16 qs[4][VJ * HD], ks[4][VJ * HD], vs[4][VJ * HD], es[4][VJ * HD];
    __shared__ float    rps[4][VJ * HD];   // n_hop*32 <= 800
    __shared__ float    S[4][VJ * VJ];
    __shared__ float    dd[4][VJ];
    int w = threadIdx.x >> 5, lane = threadIdx.x & 31;
    int u = blockIdx.x * 4 + w;            // (b*TT+t)*HH + h
    int h = u % HH, bt = u / HH;
    size_t base = (size_t)bt * VJ;
    float alpha = alpha_p[0];
    for (int idx = lane; idx < VJ * HD; idx += 32) {
        int i = idx >> 5, c = idx & 31;
        size_t r = base + i;
        qs[w][idx] = qb[r * DIMC + h * HD + c];
        ks[w][idx] = kvb[r * (2 * DIMC) + h * HD + c];
        vs[w][idx] = kvb[r * (2 * DIMC) + DIMC + h * HD + c];
        es[w][idx] = eb[r * DIMC + h * HD + c];
    }
    int nh = n_hop > VJ ? VJ : n_hop;
    for (int idx = lane; idx < nh * HD; idx += 32)
        rps[w][idx] = rpe[(idx >> 5) * DIMC + h * HD + (idx & 31)];
    __syncthreads();
    if (lane < VJ) {
        float a = 0.0f;
        #pragma unroll
        for (int c = 0; c < HD; ++c) a += w1[h * HD + c] * (float)es[w][lane * HD + c];
        dd[w][lane] = a;
    }
    __syncthreads();
    for (int idx = lane; idx < VJ * VJ; idx += 32) {
        int i = idx / VJ, j = idx % VJ;
        int hop = hops[i * VJ + j];
        const float* rp = &rps[w][hop * HD];
        float a = dd[w][j];
        #pragma unroll
        for (int c = 0; c < HD; ++c) {
            float qv = (float)qs[w][i * HD + c];
            a += qv * ((float)ks[w][j * HD + c] + (float)es[w][j * HD + c] + rp[c]);
        }
        S[w][idx] = a * SCALE_F;
    }
    __syncthreads();
    if (lane < VJ) {
        float m = -1e30f;
        for (int j = 0; j < VJ; ++j) m = fmaxf(m, S[w][lane * VJ + j]);
        float ex[VJ], sum = 0.0f;
        for (int j = 0; j < VJ; ++j) { ex[j] = expf(S[w][lane * VJ + j] - m); sum += ex[j]; }
        float inv = 1.0f / sum;
        for (int j = 0; j < VJ; ++j)
            S[w][lane * VJ + j] = alpha * ex[j] * inv + outer[h * VJ * VJ + lane * VJ + j];
    }
    __syncthreads();
    for (int idx = lane; idx < VJ * HD; idx += 32) {
        int i = idx >> 5, c = idx & 31;
        float a = 0.0f;
        #pragma unroll
        for (int j = 0; j < VJ; ++j) a += S[w][i * VJ + j] * (float)vs[w][j * HD + c];
        ob[(base + i) * DIMC + h * HD + c] = (_Float16)a;
    }
}

// out[n,ch,t,v] = x[n,ch,t,v] + proj[(n,t,v),ch] + proj_b[ch]
__global__ void final_kernel(const float* __restrict__ x, const float* __restrict__ proj,
                             const float* __restrict__ pb, float* __restrict__ out, size_t total) {
    size_t i = (size_t)blockIdx.x * blockDim.x + threadIdx.x;
    if (i >= total) return;
    int v = (int)(i % VJ);
    size_t r = i / VJ;
    int t = (int)(r % TT); r /= TT;
    int ch = (int)(r % DIMC);
    int n = (int)(r / DIMC);
    size_t row = ((size_t)n * TT + t) * VJ + v;
    out[i] = x[i] + proj[row * DIMC + ch] + pb[ch];
}

// ---------------- launcher ----------------

extern "C" void kernel_launch(void* const* d_in, const int* in_sizes, int n_in,
                              void* d_out, int out_size, void* d_ws, size_t ws_size,
                              hipStream_t stream) {
    const float* x       = (const float*)d_in[0];
    const int*   jlabel  = (const int*)d_in[1];
    const float* he_w    = (const float*)d_in[3];
    const int*   hops    = (const int*)d_in[4];
    const float* ln_g    = (const float*)d_in[5];
    const float* ln_b    = (const float*)d_in[6];
    const float* hgc_w   = (const float*)d_in[7];
    const float* q_w     = (const float*)d_in[8];
    const float* kv_w    = (const float*)d_in[9];
    const float* proj_w  = (const float*)d_in[10];
    const float* proj_b  = (const float*)d_in[11];
    const float* rpe     = (const float*)d_in[12];
    const float* w1      = (const float*)d_in[13];
    const float* outer   = (const float*)d_in[14];
    const float* alpha   = (const float*)d_in[15];
    const float* lin1_w  = (const float*)d_in[16];
    const float* lin1_b  = (const float*)d_in[17];
    const float* lin2_w  = (const float*)d_in[18];
    const float* lin2_b  = (const float*)d_in[19];
    int n_hop = in_sizes[12] / DIMC;

    // workspace carve-up (256B aligned)
    char* p = (char*)d_ws;
    auto carve = [&](size_t bytes) { char* r = p; p += (bytes + 255) & ~(size_t)255; return r; };
    const size_t R = ROWS;
    _Float16* zbuf   = (_Float16*)carve(R * DIMC * 2);
    _Float16* xnbuf  = (_Float16*)carve(R * DIMC * 2);
    float*    opbuf  = (float*)   carve(R * DIMC * 4);
    _Float16* qbuf   = (_Float16*)carve(R * DIMC * 2);
    _Float16* kvbuf  = (_Float16*)carve(R * 2 * DIMC * 2);
    _Float16* ebuf   = (_Float16*)carve(R * DIMC * 2);
    _Float16* obuf   = (_Float16*)carve(R * DIMC * 2);
    float*    projbf = (float*)   carve(R * DIMC * 4);
    float*    h_adj  = (float*)   carve(VJ * VJ * 4);
    _Float16* whgc   = (_Float16*)carve(DIMC * DIMC * 2);
    _Float16* wq     = (_Float16*)carve(DIMC * DIMC * 2);
    _Float16* wkv    = (_Float16*)carve(2 * DIMC * DIMC * 2);
    _Float16* wbd    = (_Float16*)carve(DIMC * DIMC * 2);
    float*    sbuf   = (float*)   carve(NB * TT * 4);
    float*    gbuf   = (float*)   carve(NB * TT * 4);

    // 1. prep
    prep_hadj_kernel<<<1, 256, 0, stream>>>(jlabel, he_w, h_adj);
    f2h_kernel<<<(DIMC * DIMC + 255) / 256, 256, 0, stream>>>(hgc_w, whgc, DIMC * DIMC);
    f2h_kernel<<<(DIMC * DIMC + 255) / 256, 256, 0, stream>>>(q_w, wq, DIMC * DIMC);
    f2h_kernel<<<(2 * DIMC * DIMC + 255) / 256, 256, 0, stream>>>(kv_w, wkv, 2 * DIMC * DIMC);
    wbd_kernel<<<(DIMC * DIMC + 255) / 256, 256, 0, stream>>>(proj_w, wbd);

    // 2. hypergraph mix -> z (f16 rows)
    z_kernel<<<NB * TT, 256, 0, stream>>>(x, h_adj, zbuf);

    // 3. op = z @ hgc_w^T  (f32)
    {
        int waves = (R / 16) * (DIMC / 16);
        gemm_wmma_kernel<<<waves / 8, 256, 0, stream>>>(zbuf, whgc, opbuf, nullptr,
                                                        (int)R, DIMC, DIMC, 0);
    }

    // 4. LayerNorm -> xn (f16 rows)
    ln_kernel<<<NB * TT, 256, 0, stream>>>(x, ln_g, ln_b, xnbuf);

    // 5. q / kv projections (f16 out)
    {
        int waves_q = (R / 16) * (DIMC / 16);
        gemm_wmma_kernel<<<waves_q / 8, 256, 0, stream>>>(xnbuf, wq, nullptr, qbuf,
                                                          (int)R, DIMC, DIMC, 1);
        int waves_kv = (R / 16) * (2 * DIMC / 16);
        gemm_wmma_kernel<<<waves_kv / 8, 256, 0, stream>>>(xnbuf, wkv, nullptr, kvbuf,
                                                           (int)R, DIMC, 2 * DIMC, 1);
    }

    // 6. squeeze-excite gate, e branch
    s_kernel<<<NB * TT, 256, 0, stream>>>(opbuf, sbuf);
    g_kernel<<<NB, TT, 0, stream>>>(sbuf, lin1_w, lin1_b, lin2_w, lin2_b, gbuf);
    {
        size_t total = R * DIMC;
        e_kernel<<<(int)((total + 255) / 256), 256, 0, stream>>>(opbuf, gbuf, ebuf, total);
    }

    // 7. attention (one wave per (b,t,h))
    {
        int units = NB * TT * HH;          // 24576
        attn_kernel<<<units / 4, 128, 0, stream>>>(qbuf, kvbuf, ebuf, hops, rpe, w1,
                                                   outer, alpha, obuf, n_hop);
    }

    // 8. grouped projection via block-diagonal GEMM (f32 out)
    {
        int waves = (R / 16) * (DIMC / 16);
        gemm_wmma_kernel<<<waves / 8, 256, 0, stream>>>(obuf, wbd, projbf, nullptr,
                                                        (int)R, DIMC, DIMC, 0);
    }

    // 9. residual + bias, scatter back to (n,c,t,v)
    {
        size_t total = (size_t)NB * DIMC * TT * VJ;
        final_kernel<<<(int)((total + 255) / 256), 256, 0, stream>>>(x, projbf, proj_b,
                                                                     (float*)d_out, total);
    }
}
